// CoAttention_cross_74912819577425
// MI455X (gfx1250) — matrix operand order repaired and checked by reading, not compiled, not run on hardware
//
#include <hip/hip_runtime.h>

// Problem constants (from reference): B=4, S=2048, H=1024, NH=16, D=64
#define BB 4
#define SS 2048
#define HH 1024
#define NHH 16
#define DD 64

typedef __attribute__((ext_vector_type(16))) __bf16 v16bf;
typedef __attribute__((ext_vector_type(8)))  __bf16 v8bf;
typedef __attribute__((ext_vector_type(4)))  __bf16 v4bf;
typedef __attribute__((ext_vector_type(8)))  float  v8f;
typedef __attribute__((ext_vector_type(4)))  float  v4f;

static __device__ inline v8f wmma_bf16(v16bf a, v16bf b, v8f c) {
    // D(16x16,f32) = A(16x32,bf16) x B(32x16,bf16) + C
    return __builtin_amdgcn_wmma_f32_16x16x32_bf16(
        /*neg_a=*/false, a, /*neg_b=*/false, b,
        /*c_mod=*/(short)0, c, /*reuse_a=*/false, /*reuse_b=*/false);
}

static __device__ inline v8f vzero8() {
    v8f z;
#pragma unroll
    for (int i = 0; i < 8; ++i) z[i] = 0.0f;
    return z;
}

// Async global->LDS copy of 32 bytes per lane (two b128 ops; INST_OFFSET
// applies to both the LDS and global addresses, so regs are shared).
static __device__ inline void async_copy_b256(void* lds_ptr, const void* gptr) {
    unsigned loff = (unsigned)(uintptr_t)lds_ptr;          // low 32 bits = LDS byte offset
    unsigned long long g = (unsigned long long)(uintptr_t)gptr;
    asm volatile("global_load_async_to_lds_b128 %0, %1, off\n\t"
                 "global_load_async_to_lds_b128 %0, %1, off offset:16"
                 :: "v"(loff), "v"(g) : "memory");
}
static __device__ inline void async_wait0() {
    asm volatile("s_wait_asynccnt 0x0" ::: "memory");
}

// ---- DPP-based 16-lane max reduction (no LDS round trips) -----------------
static constexpr unsigned dppsel(int a,int b,int c,int d,int e,int f,int g,int h){
    return (unsigned)(a|(b<<3)|(c<<6)|(d<<9)|(e<<12)|(f<<15)|(g<<18)|(h<<21));
}
template <unsigned SEL>
static __device__ inline float movdpp8f(float x) {
    return __int_as_float(__builtin_amdgcn_mov_dpp8(__float_as_int(x), SEL));
}
static __device__ inline float rowmax16(float x) {
    x = fmaxf(x, movdpp8f<dppsel(1,0,3,2,5,4,7,6)>(x));   // xor 1
    x = fmaxf(x, movdpp8f<dppsel(2,3,0,1,6,7,4,5)>(x));   // xor 2
    x = fmaxf(x, movdpp8f<dppsel(4,5,6,7,0,1,2,3)>(x));   // xor 4
    int i = __float_as_int(x);                             // merge 8-groups: DPP16 row_ror:8
    int r = __builtin_amdgcn_update_dpp(i, i, 0x128, 0xf, 0xf, true);
    return fmaxf(x, __int_as_float(r));
}

// ---------------------------------------------------------------------------
// fp32 -> bf16 bulk convert (weights), 4 elements per thread
// ---------------------------------------------------------------------------
__global__ __launch_bounds__(256)
void cvt_f32_bf16(const float* __restrict__ src, __bf16* __restrict__ dst)
{
    const int i = (blockIdx.x * 256 + threadIdx.x) * 4;
    v4f f = *(const v4f*)(src + i);
    v4bf o;
#pragma unroll
    for (int j = 0; j < 4; ++j) o[j] = (__bf16)f[j];
    *(v4bf*)(dst + i) = o;
}

// ---------------------------------------------------------------------------
// GEMM:  out[m,n] = bf16( sum_k A[m,k] * Wb[n,k] + bias[n] )
// A is [M,K] (fp32 if A_F32 else bf16), Wb is [N,K] bf16, out is [M,N] bf16.
// Workgroup tile 128x128, 8 waves (4x2), each wave 2x4 of 16x16 WMMA tiles.
// Double-buffered LDS staging via GLOBAL_LOAD_ASYNC_TO_LDS_B128 (ASYNCcnt).
// ---------------------------------------------------------------------------
template <bool A_F32>
__global__ __launch_bounds__(256)
void gemm_nt_bias_bf16(const void* __restrict__ Araw,
                       const __bf16* __restrict__ Wb,
                       const float* __restrict__ bias,
                       __bf16* __restrict__ out,
                       int M, int N, int K)
{
    __shared__ __bf16 Al[2][128][40];   // 128 x 32 (+8 pad), row stride 80B
    __shared__ __bf16 Bl[2][128][40];

    const int tid  = threadIdx.x;
    const int lane = tid & 31;
    const int wave = tid >> 5;
    const int wm   = wave & 3;   // wave M position -> +wm*32
    const int wn   = wave >> 2;  // wave N position -> +wn*64
    const int Mb   = blockIdx.x * 128;
    const int Nb   = blockIdx.y * 128;
    const int l16  = lane & 15;
    const int hiH  = (lane >= 16);

    // cooperative loader: thread -> (row 0..127, 16 contiguous K elements)
    const int lrow  = tid >> 1;
    const int lkoff = (tid & 1) * 16;

    auto stage = [&](int k0, int buf) {
        if (A_F32) {
            const float* A = (const float*)Araw;
            const float* src = A + (size_t)(Mb + lrow) * K + k0 + lkoff;
            v4f f0 = ((const v4f*)src)[0];
            v4f f1 = ((const v4f*)src)[1];
            v4f f2 = ((const v4f*)src)[2];
            v4f f3 = ((const v4f*)src)[3];
            v8bf lo, hi;
#pragma unroll
            for (int i = 0; i < 4; ++i) {
                lo[i]     = (__bf16)f0[i];
                lo[4 + i] = (__bf16)f1[i];
                hi[i]     = (__bf16)f2[i];
                hi[4 + i] = (__bf16)f3[i];
            }
            *(v8bf*)&Al[buf][lrow][lkoff]     = lo;
            *(v8bf*)&Al[buf][lrow][lkoff + 8] = hi;
        } else {
            const __bf16* A = (const __bf16*)Araw;
            async_copy_b256(&Al[buf][lrow][lkoff],
                            A + (size_t)(Mb + lrow) * K + k0 + lkoff);
        }
        async_copy_b256(&Bl[buf][lrow][lkoff],
                        Wb + (size_t)(Nb + lrow) * K + k0 + lkoff);
    };

    v8f acc[2][4];
#pragma unroll
    for (int mi = 0; mi < 2; ++mi)
#pragma unroll
        for (int ni = 0; ni < 4; ++ni) acc[mi][ni] = vzero8();

    const int nsteps = K / 32;
    stage(0, 0);
    for (int it = 0; it < nsteps; ++it) {
        const int cur = it & 1;
        async_wait0();          // this wave's copies into `cur` complete
        __syncthreads();        // all waves' copies complete / prev reads done
        if (it + 1 < nsteps) stage((it + 1) * 32, cur ^ 1);   // overlap next DMA

        // ---- fragments + WMMA (from buffer `cur`) ----
        v16bf af[2];
#pragma unroll
        for (int mi = 0; mi < 2; ++mi) {
            const int m  = wm * 32 + mi * 16 + l16;
            const int ak = hiH ? 8 : 0;   // A: lanes 0-15 K{0-7,16-23}; 16-31 K{8-15,24-31}
            v8bf lo = *(const v8bf*)&Al[cur][m][ak];
            v8bf hi = *(const v8bf*)&Al[cur][m][ak + 16];
#pragma unroll
            for (int i = 0; i < 8; ++i) { af[mi][i] = lo[i]; af[mi][8 + i] = hi[i]; }
        }
        v16bf bfr[4];
#pragma unroll
        for (int ni = 0; ni < 4; ++ni) {
            const int n  = wn * 64 + ni * 16 + l16;
            const int bk = hiH ? 8 : 0;   // B: lanes 0-15 K{0-7,16-23}; 16-31 K{8-15,24-31}
            v8bf lo = *(const v8bf*)&Bl[cur][n][bk];
            v8bf hi = *(const v8bf*)&Bl[cur][n][bk + 16];
#pragma unroll
            for (int i = 0; i < 8; ++i) { bfr[ni][i] = lo[i]; bfr[ni][8 + i] = hi[i]; }
        }
#pragma unroll
        for (int mi = 0; mi < 2; ++mi)
#pragma unroll
            for (int ni = 0; ni < 4; ++ni)
                acc[mi][ni] = wmma_bf16(af[mi], bfr[ni], acc[mi][ni]);
    }

    // ---- epilogue: bias + bf16 store ----
#pragma unroll
    for (int mi = 0; mi < 2; ++mi) {
#pragma unroll
        for (int ni = 0; ni < 4; ++ni) {
            const int col   = Nb + wn * 64 + ni * 16 + l16;
            const float bv  = bias[col];
            const int rbase = Mb + wm * 32 + mi * 16 + (hiH ? 8 : 0);
#pragma unroll
            for (int r = 0; r < 8; ++r)
                out[(size_t)(rbase + r) * N + col] = (__bf16)(acc[mi][ni][r] + bv);
        }
    }
}

// ---------------------------------------------------------------------------
// Flash attention: one (b,h) per block; 8 waves, each owning 16 queries.
// KV streamed in 64-token chunks, double-buffered: K via async global->LDS
// DMA, V staged transposed. Row-sum via WMMA (P x ones); running max via
// DPP butterflies. All matmuls via WMMA.
// ---------------------------------------------------------------------------
__global__ __launch_bounds__(256)
void attn_flash(const __bf16* __restrict__ Qb,
                const __bf16* __restrict__ Kb,
                const __bf16* __restrict__ Vb,
                const float* __restrict__ mask,
                float* __restrict__ out)
{
    __shared__ __bf16 Kl[2][64][72];     // 64 tokens x 64 dims (+8 pad)
    __shared__ __bf16 Vt[2][64][72];     // transposed: dim x 64 tokens (+8 pad)
    __shared__ __bf16 Pl[8][16][72];     // per-wave P transpose scratch (16 x 64)

    const int tid  = threadIdx.x;
    const int lane = tid & 31;
    const int wave = tid >> 5;
    const int b    = blockIdx.z;
    const int h    = blockIdx.y;
    const int q0   = (blockIdx.x * 8 + wave) * 16;
    const int l16  = lane & 15;
    const int hiH  = (lane >= 16);
    const float scale = 0.125f;          // 1/sqrt(64)

    auto stageKV = [&](int kb, int buf) {
        const int tok = tid >> 2;             // 0..63
        const int dp  = (tid & 3) * 16;       // 0,16,32,48
        const size_t base = ((size_t)(b * SS + kb + tok)) * HH + h * DD + dp;
        async_copy_b256(&Kl[buf][tok][dp], Kb + base);
        v8bf v0 = *(const v8bf*)(Vb + base);
        v8bf v1 = *(const v8bf*)(Vb + base + 8);
#pragma unroll
        for (int i = 0; i < 8; ++i) {
            Vt[buf][dp + i][tok]     = v0[i];
            Vt[buf][dp + 8 + i][tok] = v1[i];
        }
    };

    // ---- Q fragments: A(16x32) x2 covering d=0..31, 32..63 ----
    v16bf qf[2];
    {
        const __bf16* qrow = Qb + ((size_t)(b * SS + q0 + l16)) * HH + h * DD;
#pragma unroll
        for (int j = 0; j < 2; ++j) {
            const int k0 = j * 32 + (hiH ? 8 : 0);
            v8bf lo = *(const v8bf*)(qrow + k0);
            v8bf hi = *(const v8bf*)(qrow + k0 + 16);
#pragma unroll
            for (int i = 0; i < 8; ++i) { qf[j][i] = lo[i]; qf[j][8 + i] = hi[i]; }
        }
    }
    // all-ones B fragment for WMMA row-sum
    v16bf onesB;
#pragma unroll
    for (int i = 0; i < 16; ++i) onesB[i] = (__bf16)1.0f;

    v8f O[4];
#pragma unroll
    for (int dc = 0; dc < 4; ++dc) O[dc] = vzero8();
    float mrow[8], lrow[8];
#pragma unroll
    for (int r = 0; r < 8; ++r) { mrow[r] = -3.0e38f; lrow[r] = 0.0f; }

    const int nchunks = SS / 64;
    stageKV(0, 0);
    for (int it = 0; it < nchunks; ++it) {
        const int cur = it & 1;
        const int kb  = it * 64;
        async_wait0();
        __syncthreads();
        if (it + 1 < nchunks) stageKV(kb + 64, cur ^ 1);   // overlap next chunk DMA

        // ---- scores: four 16x16 tiles (keys kb+0..63) ----
        v8f s[4];
#pragma unroll
        for (int t = 0; t < 4; ++t) {
            v8f sc = vzero8();
#pragma unroll
            for (int j = 0; j < 2; ++j) {
                const int kr = t * 16 + l16;                 // B column = key token
                const int dk = j * 32 + (hiH ? 8 : 0);       // reduction dim (=d)
                v8bf lo = *(const v8bf*)&Kl[cur][kr][dk];
                v8bf hi = *(const v8bf*)&Kl[cur][kr][dk + 16];
                v16bf kf;
#pragma unroll
                for (int i = 0; i < 8; ++i) { kf[i] = lo[i]; kf[8 + i] = hi[i]; }
                sc = wmma_bf16(qf[j], kf, sc);
            }
            s[t] = sc;
        }

        // ---- online softmax: DPP butterflies for the running max ----
        float mk[4];
#pragma unroll
        for (int t = 0; t < 4; ++t)
            mk[t] = mask[(size_t)b * SS + kb + t * 16 + l16];
        float corr[8];
#pragma unroll
        for (int r = 0; r < 8; ++r) {
            float a0 = s[0][r] * scale + mk[0];
            float a1 = s[1][r] * scale + mk[1];
            float a2 = s[2][r] * scale + mk[2];
            float a3 = s[3][r] * scale + mk[3];
            const float mx = rowmax16(fmaxf(fmaxf(a0, a1), fmaxf(a2, a3)));
            const float mn = fmaxf(mrow[r], mx);
            s[0][r] = __expf(a0 - mn);
            s[1][r] = __expf(a1 - mn);
            s[2][r] = __expf(a2 - mn);
            s[3][r] = __expf(a3 - mn);
            corr[r] = __expf(mrow[r] - mn);
            mrow[r] = mn;
        }
#pragma unroll
        for (int dc = 0; dc < 4; ++dc)
#pragma unroll
            for (int r = 0; r < 8; ++r) O[dc][r] *= corr[r];

        // ---- transpose P through LDS into A-fragment layout ----
        {
            const int prow = hiH ? 8 : 0;
#pragma unroll
            for (int t = 0; t < 4; ++t)
#pragma unroll
                for (int r = 0; r < 8; ++r)
                    Pl[wave][prow + r][t * 16 + l16] = (__bf16)s[t][r];
        }
        v16bf pf[2];
#pragma unroll
        for (int j = 0; j < 2; ++j) {
            const int pk = j * 32 + (hiH ? 8 : 0);
            v8bf lo = *(const v8bf*)&Pl[wave][l16][pk];
            v8bf hi = *(const v8bf*)&Pl[wave][l16][pk + 16];
#pragma unroll
            for (int i = 0; i < 8; ++i) { pf[j][i] = lo[i]; pf[j][8 + i] = hi[i]; }
        }

        // ---- row-sum via WMMA: rowsum(P) = P x ones ----
        {
            v8f rs = vzero8();
            rs = wmma_bf16(pf[0], onesB, rs);
            rs = wmma_bf16(pf[1], onesB, rs);
#pragma unroll
            for (int r = 0; r < 8; ++r)
                lrow[r] = lrow[r] * corr[r] + rs[r];
        }

        // ---- O += P x V ----
#pragma unroll
        for (int j = 0; j < 2; ++j) {
            const int tk = j * 32 + (hiH ? 8 : 0);       // reduction dim = token
#pragma unroll
            for (int dc = 0; dc < 4; ++dc) {
                const int n = dc * 16 + l16;             // B column = output dim
                v8bf lo = *(const v8bf*)&Vt[cur][n][tk];
                v8bf hi = *(const v8bf*)&Vt[cur][n][tk + 16];
                v16bf vf;
#pragma unroll
                for (int i = 0; i < 8; ++i) { vf[i] = lo[i]; vf[8 + i] = hi[i]; }
                O[dc] = wmma_bf16(pf[j], vf, O[dc]);
            }
        }
    }

    // ---- normalize + store fp32 ctx in [B,S,H] layout ----
#pragma unroll
    for (int dc = 0; dc < 4; ++dc) {
        const int col   = h * DD + dc * 16 + l16;
        const int rbase = q0 + (hiH ? 8 : 0);
#pragma unroll
        for (int r = 0; r < 8; ++r)
            out[((size_t)b * SS + rbase + r) * HH + col] = O[dc][r] / lrow[r];
    }
}

// ---------------------------------------------------------------------------
extern "C" void kernel_launch(void* const* d_in, const int* in_sizes, int n_in,
                              void* d_out, int out_size, void* d_ws, size_t ws_size,
                              hipStream_t stream)
{
    const float* X    = (const float*)d_in[0];  // hidden_states [B,S,H]
    const float* mask = (const float*)d_in[1];  // attention_mask [B,1,1,S]
    const float* Wq   = (const float*)d_in[2];
    const float* bq   = (const float*)d_in[3];
    const float* Wk   = (const float*)d_in[4];
    const float* bk   = (const float*)d_in[5];
    const float* Wv   = (const float*)d_in[6];
    const float* bv   = (const float*)d_in[7];
    float* out = (float*)d_out;

    // Workspace layout (bf16): mixed_q/K/V (16 MiB each) + Wq/Wk/Wv (2 MiB each)
    const size_t NTOK = (size_t)BB * SS;        // 8192
    __bf16* Qb  = (__bf16*)d_ws;
    __bf16* Kb  = Qb  + NTOK * HH;
    __bf16* Vb  = Kb  + NTOK * HH;
    __bf16* Wqb = Vb  + NTOK * HH;
    __bf16* Wkb = Wqb + (size_t)HH * HH;
    __bf16* Wvb = Wkb + (size_t)HH * HH;

    // Pre-convert weights fp32 -> bf16 (1M elements each; 4 per thread)
    dim3 cGrid((HH * HH) / (256 * 4));
    cvt_f32_bf16<<<cGrid, dim3(256), 0, stream>>>(Wq, Wqb);
    cvt_f32_bf16<<<cGrid, dim3(256), 0, stream>>>(Wk, Wkb);
    cvt_f32_bf16<<<cGrid, dim3(256), 0, stream>>>(Wv, Wvb);

    dim3 gGrid(64, 8);   // M/128 x N/128 for M=8192, N=1024
    dim3 gBlk(256);
    // mixed_q = X @ Wq^T + bq
    gemm_nt_bias_bf16<true ><<<gGrid, gBlk, 0, stream>>>(X,  Wqb, bq, Qb, 8192, 1024, 1024);
    // k = mixed_q @ Wk^T + bk ; v = mixed_q @ Wv^T + bv  (source quirk: both from mixed_q)
    gemm_nt_bias_bf16<false><<<gGrid, gBlk, 0, stream>>>(Qb, Wkb, bk, Kb, 8192, 1024, 1024);
    gemm_nt_bias_bf16<false><<<gGrid, gBlk, 0, stream>>>(Qb, Wvb, bv, Vb, 8192, 1024, 1024);

    dim3 aGrid(SS / 128, NHH, BB);  // (16,16,4): 8 query tiles of 16 per block
    attn_flash<<<aGrid, dim3(256), 0, stream>>>(Qb, Kb, Vb, mask, out);
}
